// SpatialGATLayer_69200513073693
// MI455X (gfx1250) — matrix operand order repaired and checked by reading, not compiled, not run on hardware
//
#include <hip/hip_runtime.h>

typedef _Float16 h16;
typedef __attribute__((ext_vector_type(16))) _Float16 v16h;
typedef __attribute__((ext_vector_type(8)))  _Float16 v8h;
typedef __attribute__((ext_vector_type(8)))  float    v8f;
typedef __attribute__((ext_vector_type(4)))  unsigned int u32x4;
typedef __attribute__((ext_vector_type(8)))  int          i32x8;
typedef __attribute__((ext_vector_type(4)))  int          i32x4;

static constexpr int       Dd   = 128;
static constexpr int       FFd  = 2048;
static constexpr int       Eg   = 7000;
static constexpr int       Nn   = 883;
static constexpr long long TNc  = 8LL * 12 * 883;   // 84768 nodes
static constexpr long long EEg  = 8LL * 12 * 7000;  // 672000 graph edges
static constexpr long long ETOT = EEg + TNc;        // + self loops = 756768

// ---------------- WMMA helpers (CDNA5 16x16x32 f16 -> f32) ----------------

__device__ __forceinline__ v16h make_frag(const h16* p, int gap) {
  // lane-private fragment: two contiguous 8xf16 chunks (16B loads)
  v8h lo = *(const v8h*)p;
  v8h hi = *(const v8h*)(p + gap);
  v16h r;
#pragma unroll
  for (int i = 0; i < 8; ++i) { r[i] = lo[i]; r[i + 8] = hi[i]; }
  return r;
}

__device__ __forceinline__ v8f wmma_f16(v16h a, v16h b, v8f c) {
  return __builtin_amdgcn_wmma_f32_16x16x32_f16(false, a, false, b, (short)0, c,
                                                false, false);
}

// monotonic float<->u32 encoding for atomicMax-based segment max
__device__ __forceinline__ unsigned enc_f(float f) {
  unsigned u = __float_as_uint(f);
  return (u & 0x80000000u) ? ~u : (u | 0x80000000u);
}
__device__ __forceinline__ float dec_f(unsigned u) {
  unsigned b = (u & 0x80000000u) ? (u & 0x7fffffffu) : ~u;
  return __uint_as_float(b);
}

__device__ __forceinline__ void edge_sd(long long e, const long long* ei,
                                        long long& src, long long& dst) {
  if (e < EEg) {
    long long inst = e / Eg;
    long long k = e - inst * Eg;
    src = ei[k] + inst * Nn;
    dst = ei[Eg + k] + inst * Nn;
  } else {
    src = dst = e - EEg;  // self loop
  }
}

// ---------------- small utility kernels ----------------

__global__ void k_fill0(unsigned* p, long long n) {
  long long i = (long long)blockIdx.x * blockDim.x + threadIdx.x;
  if (i < n) p[i] = 0u;
}

__global__ void k_f2h(const float* __restrict__ x, h16* __restrict__ y, long long n) {
  long long i = (long long)blockIdx.x * blockDim.x + threadIdx.x;
  if (i < n) y[i] = (h16)x[i];
}

// W [K,N] f32 -> Wt [N,K] f16
__global__ void k_transpose_h(const float* __restrict__ W, h16* __restrict__ Wt,
                              int K, int N) {
  long long i = (long long)blockIdx.x * blockDim.x + threadIdx.x;
  if (i >= (long long)K * N) return;
  int k = (int)(i % K);
  int n = (int)(i / K);
  Wt[(long long)n * K + k] = (h16)W[(long long)k * N + n];
}

// ---------------- WMMA GEMM: C[M,N] = A[M,K] @ B[K,N] + bias ----------------
// A row-major f16; Bt = B^T row-major f16 [N,K]. Block: 256 thr = 8 waves,
// tile 64(M) x 128(N); wave -> 16x64 via 4 accumulators. Software-pipelined:
// fragments for step k+32 are loaded while WMMAs for step k issue.
__global__ void k_gemm(const h16* __restrict__ A, const h16* __restrict__ Bt,
                       const float* __restrict__ bias, float* __restrict__ C,
                       int M, int N, int K) {
  const int lane = threadIdx.x & 31;
  const int w = threadIdx.x >> 5;
  const int wm = w & 3, wn = w >> 2;
  const int mbase = blockIdx.x * 64 + wm * 16;
  const int nbase = blockIdx.y * 128 + wn * 64;

  int arow = mbase + (lane & 15);
  if (arow >= M) arow = M - 1;  // clamp loads; stores guarded below
  const int kbA = (lane >> 4) * 16;
  const int kbB = (lane >> 4) * 16;
  const int col0 = nbase + (lane & 15);

  const h16* pa = A + (long long)arow * K + kbA;
  const h16* pb[4];
#pragma unroll
  for (int j = 0; j < 4; ++j) pb[j] = Bt + (long long)(col0 + j * 16) * K + kbB;

  v8f acc[4];
#pragma unroll
  for (int j = 0; j < 4; ++j)
#pragma unroll
    for (int i = 0; i < 8; ++i) acc[j][i] = 0.f;

  v16h a = make_frag(pa, 8);
  v16h b[4];
#pragma unroll
  for (int j = 0; j < 4; ++j) b[j] = make_frag(pb[j], 8);

  for (int k = 0; k < K; k += 32) {
    v16h an = a;
    v16h bn[4];
#pragma unroll
    for (int j = 0; j < 4; ++j) bn[j] = b[j];
    if (k + 32 < K) {
      an = make_frag(pa + k + 32, 8);
#pragma unroll
      for (int j = 0; j < 4; ++j) bn[j] = make_frag(pb[j] + k + 32, 8);
    }
#pragma unroll
    for (int j = 0; j < 4; ++j) acc[j] = wmma_f16(a, b[j], acc[j]);
    a = an;
#pragma unroll
    for (int j = 0; j < 4; ++j) b[j] = bn[j];
  }

  const int rbase = mbase + (lane >> 4) * 8;
#pragma unroll
  for (int j = 0; j < 4; ++j) {
    const int col = col0 + j * 16;
    const float bc = bias ? bias[col] : 0.f;
#pragma unroll
    for (int i = 0; i < 8; ++i) {
      const int r = rbase + i;
      if (r < M) C[(long long)r * N + col] = acc[j][i] + bc;
    }
  }
}

// ---------------- edge attention kernels ----------------

// 4 waves per edge (128 threads/edge): score[e,h] = sum_c leaky(xl[s]+xr[d]) * att
__global__ void k_score(const long long* __restrict__ ei, const float* __restrict__ xl,
                        const float* __restrict__ xr, const float* __restrict__ att,
                        float* __restrict__ score, unsigned* __restrict__ smax) {
  long long gid = (long long)blockIdx.x * 256 + threadIdx.x;
  long long e = gid >> 7;
  int c = (int)(gid & 127);
  if (e >= ETOT) return;
  long long src, dst;
  edge_sd(e, ei, src, dst);
  float t = xl[src * Dd + c] + xr[dst * Dd + c];
  t = t > 0.f ? t : 0.2f * t;
  float p = t * att[c];
#pragma unroll
  for (int m = 8; m >= 1; m >>= 1) p += __shfl_xor(p, m, 32);
  if ((c & 15) == 0) {
    int h = c >> 4;
    score[e * 8 + h] = p;
    atomicMax(&smax[dst * 8 + h], enc_f(p));
  }
}

__global__ void k_expsum(const long long* __restrict__ ei, float* __restrict__ score,
                         const unsigned* __restrict__ smax, float* __restrict__ denom) {
  long long gid = (long long)blockIdx.x * 256 + threadIdx.x;
  if (gid >= ETOT * 8) return;
  long long e = gid >> 3;
  int h = (int)(gid & 7);
  long long src, dst;
  edge_sd(e, ei, src, dst);
  (void)src;
  float v = __expf(score[gid] - dec_f(smax[dst * 8 + h]));
  score[gid] = v;  // reuse buffer as exp(score - max)
  atomicAdd(&denom[dst * 8 + h], v);
}

__global__ void k_agg(const long long* __restrict__ ei, const float* __restrict__ a,
                      const float* __restrict__ denom, const float* __restrict__ xl,
                      float* __restrict__ agg) {
  long long gid = (long long)blockIdx.x * 256 + threadIdx.x;
  long long e = gid >> 7;
  int c = (int)(gid & 127);
  if (e >= ETOT) return;
  long long src, dst;
  edge_sd(e, ei, src, dst);
  int h = c >> 4;
  float alpha = a[e * 8 + h] / denom[dst * 8 + h];
  atomicAdd(&agg[dst * Dd + c], alpha * xl[src * Dd + c]);
}

// ---------------- LayerNorm (wave per 128-wide row) ----------------

__global__ void k_ln(const float* __restrict__ a, const float* __restrict__ b,
                     const float* __restrict__ bias, const float* __restrict__ g,
                     const float* __restrict__ beta, float* __restrict__ of,
                     h16* __restrict__ oh, long long rows) {
  const int lane = threadIdx.x & 31;
  long long row = (long long)blockIdx.x * 8 + (threadIdx.x >> 5);
  if (row >= rows) return;
  const float* pa = a + row * Dd;
  const float* pb = b + row * Dd;
  float v[4];
  float s = 0.f;
#pragma unroll
  for (int i = 0; i < 4; ++i) {
    int c = i * 32 + lane;
    float t = pa[c] + pb[c];
    if (bias) t += bias[c];
    v[i] = t;
    s += t;
  }
#pragma unroll
  for (int m = 16; m >= 1; m >>= 1) s += __shfl_xor(s, m, 32);
  float mu = s * (1.f / 128.f);
  float q = 0.f;
#pragma unroll
  for (int i = 0; i < 4; ++i) {
    float d = v[i] - mu;
    q += d * d;
  }
#pragma unroll
  for (int m = 16; m >= 1; m >>= 1) q += __shfl_xor(q, m, 32);
  float rstd = rsqrtf(q * (1.f / 128.f) + 1e-5f);
#pragma unroll
  for (int i = 0; i < 4; ++i) {
    int c = i * 32 + lane;
    float y = (v[i] - mu) * rstd * g[c] + beta[c];
    of[row * Dd + c] = y;
    if (oh) oh[row * Dd + c] = (h16)y;
  }
}

// ---------------- fused FFN: ff = relu(h@W1+b1)@W2 + b2 ----------------
// 64-row tile per block; 2048-wide intermediate staged in LDS as f16 in
// 128-col chunks; both GEMMs via WMMA. LDS: 2 * 64*128*2B = 32 KB.
// h tile staged to LDS via the Tensor Data Mover (one DMA per block).
__global__ void k_ffn(const h16* __restrict__ Hh, const h16* __restrict__ W1t,
                      const float* __restrict__ b1, const h16* __restrict__ W2t,
                      const float* __restrict__ b2, float* __restrict__ FFout) {
  __shared__ h16 hs[64][128];
  __shared__ h16 acts[64][128];
  const int tid = threadIdx.x;
  const int lane = tid & 31;
  const int w = tid >> 5, wm = w & 3, wn = w >> 2;
  const long long tile = blockIdx.x;

  const long long rem = TNc * 128 - tile * 8192;
  const unsigned nelem = rem < 8192 ? (unsigned)rem : 8192u;

#if __has_builtin(__builtin_amdgcn_tensor_load_to_lds) && \
    __has_builtin(__builtin_amdgcn_s_wait_tensorcnt)
  // Zero the (rare) tail of the tile before DMA fills [0, nelem).
  if (nelem < 8192u) {
    for (int idx = tid; idx < 64 * 128; idx += 256)
      if ((unsigned)idx >= nelem) ((h16*)hs)[idx] = (h16)0.f;
  }
  if (w == 0) {
    // Tensor DMA descriptor: 1-D tile of nelem f16 elements, contiguous.
    unsigned long long ga = (unsigned long long)(const void*)(Hh + tile * 8192);
    unsigned ldsa = (unsigned)(unsigned long long)&hs[0][0];  // low 32b = LDS offset
    u32x4 g0;
    g0[0] = 1u;                                               // count=1, user mode
    g0[1] = ldsa;                                             // lds_addr
    g0[2] = (unsigned)ga;                                     // global_addr[31:0]
    g0[3] = (unsigned)((ga >> 32) & 0x01FFFFFFu) | 0x80000000u;  // addr[56:32]|type=2
    i32x8 g1v;
    g1v[0] = (int)(1u << 16);                  // workgroup_mask=0, data_size=2B
    g1v[1] = (int)((nelem & 0xFFFFu) << 16);   // tensor_dim0[15:0]
    g1v[2] = (int)(((nelem >> 16) & 0xFFFFu) | (1u << 16));  // tdim0 hi | tensor_dim1=1
    g1v[3] = (int)((nelem & 0xFFFFu) << 16);   // tensor_dim1 hi=0 | tile_dim0=nelem
    g1v[4] = 0;                                // tile_dim1=0 (1-D), tile_dim2=0
    g1v[5] = (int)nelem;                       // tensor_dim0_stride (unused in 1-D)
    g1v[6] = 0;
    g1v[7] = 0;
    i32x4 gz4;
    gz4[0] = gz4[1] = gz4[2] = gz4[3] = 0;
    i32x8 gz8;
#pragma unroll
    for (int i = 0; i < 8; ++i) gz8[i] = 0;
    // 6-arg toolchain form: (g0, g1, g2, g3, g4, cpol)
    __builtin_amdgcn_tensor_load_to_lds(g0, g1v, gz4, gz4, gz8, 0);
    __builtin_amdgcn_s_wait_tensorcnt(0);
  }
#else
  for (int idx = tid; idx < 64 * 16; idx += 256) {
    int r = idx >> 4, ch = idx & 15;
    long long gr = tile * 64 + r;
    if (gr >= TNc) gr = TNc - 1;
    *(v8h*)&hs[r][ch * 8] = *(const v8h*)(Hh + gr * 128 + ch * 8);
  }
#endif
  __syncthreads();

  const int kbA = (lane >> 4) * 16;
  const int kbB = (lane >> 4) * 16;
  const int mrow = wm * 16 + (lane & 15);
  const int col0 = wn * 64 + (lane & 15);
  const int rloc = wm * 16 + (lane >> 4) * 8;

  v8f out[4];
#pragma unroll
  for (int j = 0; j < 4; ++j)
#pragma unroll
    for (int i = 0; i < 8; ++i) out[j][i] = 0.f;

  for (int chunk = 0; chunk < FFd; chunk += 128) {
    // GEMM1 chunk: act64x128 = hs(64x128) @ W1[:, chunk:chunk+128]  (pipelined)
    v8f c[4];
#pragma unroll
    for (int j = 0; j < 4; ++j)
#pragma unroll
      for (int i = 0; i < 8; ++i) c[j][i] = 0.f;

    const h16* pw1 = W1t + (long long)(chunk + col0) * 128 + kbB;
    v16h bw[4];
#pragma unroll
    for (int j = 0; j < 4; ++j) bw[j] = make_frag(pw1 + j * (16 * 128), 8);
    for (int k = 0; k < 128; k += 32) {
      v16h a = make_frag(&hs[mrow][k + kbA], 8);
      v16h bn[4];
#pragma unroll
      for (int j = 0; j < 4; ++j) bn[j] = bw[j];
      if (k + 32 < 128) {
#pragma unroll
        for (int j = 0; j < 4; ++j)
          bn[j] = make_frag(pw1 + j * (16 * 128) + k + 32, 8);
      }
#pragma unroll
      for (int j = 0; j < 4; ++j) c[j] = wmma_f16(a, bw[j], c[j]);
#pragma unroll
      for (int j = 0; j < 4; ++j) bw[j] = bn[j];
    }
    // bias + relu -> f16 act tile in LDS (D-matrix layout scatter)
#pragma unroll
    for (int j = 0; j < 4; ++j) {
      const float bb = b1[chunk + col0 + j * 16];
#pragma unroll
      for (int i = 0; i < 8; ++i) {
        float v = c[j][i] + bb;
        v = v > 0.f ? v : 0.f;
        acts[rloc + i][col0 + j * 16] = (h16)v;
      }
    }
    __syncthreads();
    // GEMM2 chunk: out64x128 += act(64x128) @ W2[chunk:chunk+128, :] (pipelined)
    const h16* pw2 = W2t + (long long)col0 * FFd + chunk + kbB;
    v16h bw2[4];
#pragma unroll
    for (int j = 0; j < 4; ++j) bw2[j] = make_frag(pw2 + j * (16 * FFd), 8);
    for (int k = 0; k < 128; k += 32) {
      v16h a2 = make_frag(&acts[mrow][k + kbA], 8);
      v16h bn2[4];
#pragma unroll
      for (int j = 0; j < 4; ++j) bn2[j] = bw2[j];
      if (k + 32 < 128) {
#pragma unroll
        for (int j = 0; j < 4; ++j)
          bn2[j] = make_frag(pw2 + j * (16 * FFd) + k + 32, 8);
      }
#pragma unroll
      for (int j = 0; j < 4; ++j) out[j] = wmma_f16(a2, bw2[j], out[j]);
#pragma unroll
      for (int j = 0; j < 4; ++j) bw2[j] = bn2[j];
    }
    __syncthreads();
  }

#pragma unroll
  for (int j = 0; j < 4; ++j) {
    const int dcol = col0 + j * 16;
    const float bb = b2[dcol];
#pragma unroll
    for (int i = 0; i < 8; ++i) {
      long long gr = tile * 64 + rloc + i;
      if (gr < TNc) FFout[gr * 128 + dcol] = out[j][i] + bb;
    }
  }
}

// ---------------- host orchestration ----------------

#define CEILDIV(a, b) (unsigned)(((long long)(a) + (b)-1) / (b))

extern "C" void kernel_launch(void* const* d_in, const int* in_sizes, int n_in,
                              void* d_out, int out_size, void* d_ws, size_t ws_size,
                              hipStream_t stream) {
  (void)in_sizes; (void)n_in; (void)out_size; (void)ws_size;
  const float*     x        = (const float*)d_in[0];
  const long long* ei       = (const long long*)d_in[1];  // int64 [2,E]
  const float*     Wl       = (const float*)d_in[2];
  const float*     bl       = (const float*)d_in[3];
  const float*     Wr       = (const float*)d_in[4];
  const float*     br       = (const float*)d_in[5];
  const float*     att      = (const float*)d_in[6];      // [H*C]=128
  const float*     bias_gat = (const float*)d_in[7];
  const float*     W1       = (const float*)d_in[8];
  const float*     b1       = (const float*)d_in[9];
  const float*     W2       = (const float*)d_in[10];
  const float*     b2       = (const float*)d_in[11];
  const float*     g1       = (const float*)d_in[12];
  const float*     be1      = (const float*)d_in[13];
  const float*     g2       = (const float*)d_in[14];
  const float*     be2      = (const float*)d_in[15];
  float*           out      = (float*)d_out;

  char* wsp = (char*)d_ws;
  auto alloc = [&](size_t bytes) -> char* {
    char* p = wsp;
    wsp += (bytes + 255) & ~(size_t)255;
    return p;
  };
  h16*      xh    = (h16*)alloc((size_t)TNc * Dd * sizeof(h16));
  h16*      WlT   = (h16*)alloc((size_t)Dd * Dd * sizeof(h16));
  h16*      WrT   = (h16*)alloc((size_t)Dd * Dd * sizeof(h16));
  h16*      W1T   = (h16*)alloc((size_t)Dd * FFd * sizeof(h16));
  h16*      W2T   = (h16*)alloc((size_t)Dd * FFd * sizeof(h16));
  float*    xl    = (float*)alloc((size_t)TNc * Dd * sizeof(float));
  float*    xr    = (float*)alloc((size_t)TNc * Dd * sizeof(float));
  float*    score = (float*)alloc((size_t)ETOT * 8 * sizeof(float));
  unsigned* smax  = (unsigned*)alloc((size_t)TNc * 8 * sizeof(unsigned));
  float*    denom = (float*)alloc((size_t)TNc * 8 * sizeof(float));
  float*    agg   = (float*)alloc((size_t)TNc * Dd * sizeof(float));
  float*    h     = (float*)alloc((size_t)TNc * Dd * sizeof(float));
  h16*      hh    = (h16*)alloc((size_t)TNc * Dd * sizeof(h16));
  float*    ff    = (float*)alloc((size_t)TNc * Dd * sizeof(float));

  const long long nxd = TNc * Dd;

  k_f2h<<<CEILDIV(nxd, 256), 256, 0, stream>>>(x, xh, nxd);
  k_transpose_h<<<CEILDIV(Dd * Dd, 256), 256, 0, stream>>>(Wl, WlT, Dd, Dd);
  k_transpose_h<<<CEILDIV(Dd * Dd, 256), 256, 0, stream>>>(Wr, WrT, Dd, Dd);
  k_transpose_h<<<CEILDIV(Dd * FFd, 256), 256, 0, stream>>>(W1, W1T, Dd, FFd);
  k_transpose_h<<<CEILDIV((long long)FFd * Dd, 256), 256, 0, stream>>>(W2, W2T, FFd, Dd);

  k_fill0<<<CEILDIV(TNc * 8, 256), 256, 0, stream>>>(smax, TNc * 8);
  k_fill0<<<CEILDIV(TNc * 8, 256), 256, 0, stream>>>((unsigned*)denom, TNc * 8);
  k_fill0<<<CEILDIV(nxd, 256), 256, 0, stream>>>((unsigned*)agg, nxd);

  // x_l, x_r projections (WMMA)
  k_gemm<<<dim3(CEILDIV(TNc, 64), 1), 256, 0, stream>>>(xh, WlT, bl, xl,
                                                        (int)TNc, Dd, Dd);
  k_gemm<<<dim3(CEILDIV(TNc, 64), 1), 256, 0, stream>>>(xh, WrT, br, xr,
                                                        (int)TNc, Dd, Dd);

  // GATv2 edge attention: score -> segment max -> exp/sum -> weighted scatter
  k_score<<<CEILDIV(ETOT * 128, 256), 256, 0, stream>>>(ei, xl, xr, att, score, smax);
  k_expsum<<<CEILDIV(ETOT * 8, 256), 256, 0, stream>>>(ei, score, smax, denom);
  k_agg<<<CEILDIV(ETOT * 128, 256), 256, 0, stream>>>(ei, score, denom, xl, agg);

  // h = LN1(x + agg + bias_gat); also emit f16 copy for FFN
  k_ln<<<CEILDIV(TNc, 8), 256, 0, stream>>>(x, agg, bias_gat, g1, be1, h, hh, TNc);

  // fused FFN (WMMA x2, TDM-staged tile, intermediate in LDS)
  k_ffn<<<CEILDIV(TNc, 64), 256, 0, stream>>>(hh, W1T, b1, W2T, b2, ff);

  // out = LN2(h + ff)
  k_ln<<<CEILDIV(TNc, 8), 256, 0, stream>>>(h, ff, nullptr, g2, be2, out, nullptr, TNc);
}
